// OoKgDetector_31636729102421
// MI455X (gfx1250) — matrix operand order repaired and checked by reading, not compiled, not run on hardware
//
#include <hip/hip_runtime.h>
#include <hip/hip_bf16.h>
#include <cstdint>
#include <cstddef>

// ---------------------------------------------------------------------------
// MI455X (gfx1250) pipeline, all heavy GEMMs on v_wmma_f32_16x16x32_bf16.
//
//   pass 0: convert 7 weight matrices f32->bf16
//   pass 1: projections  C = bf16( A_f32 @ W^T + b )      (Q's, K banks, V^T banks)
//   pass 2 (per attention set):
//       logits  S_f32 = Q_bf16 @ Kbank^T          (WMMA, K=512, TDM A-staging)
//       softmax rows  -> P = bf16(exp(s-m)), z[row]
//       O_f32  += P @ V   split-K + global_atomic_add_f32 (WMMA, TDM A-staging)
//       out = O / z
//
// ~0.65 TFLOP total; bf16 WMMA (~5 PFLOP/s dense) => ~130us compute,
// ~2.5 GB moved => ~110us at 23.3 TB/s. bf16-A tiles staged via the Tensor
// Data Mover (double-buffered, TENSORcnt), leaving VMEM for B fragments.
// ---------------------------------------------------------------------------

typedef __bf16 bf16_t;
typedef __attribute__((ext_vector_type(16))) __bf16 v16bf;
typedef __attribute__((ext_vector_type(8)))  __bf16 v8bf;
typedef __attribute__((ext_vector_type(8)))  float  v8f;
typedef unsigned int u32x4 __attribute__((ext_vector_type(4)));
typedef int          i32x4 __attribute__((ext_vector_type(4)));
typedef int          i32x8 __attribute__((ext_vector_type(8)));

#define BT 128        // block tile rows/cols
#define KSTEP 32      // WMMA K per step (bf16)
#define LDS_STRIDE 40 // padded LDS row stride (bf16 elems): 32 data + 8 pad

__device__ __forceinline__ v16bf make16(v8bf lo, v8bf hi) {
  v16bf o;
#pragma unroll
  for (int i = 0; i < 8; ++i) { o[i] = lo[i]; o[i + 8] = hi[i]; }
  return o;
}

// A-fragment (16x32 bf16) from LDS tile. contiguous K order used here (matches frag_b_g):
// lanes 0-15: row M=lane, K = {0..15}; lanes 16-31: row M=lane-16, K={16..31}
__device__ __forceinline__ v16bf frag_a_lds(const bf16_t* lds, int rowBase, int lane) {
  int r  = rowBase + (lane & 15);
  int k0 = (lane & 16) ? 16 : 0;
  const bf16_t* p = lds + r * LDS_STRIDE + k0;
  v8bf lo = *(const v8bf*)p;
  v8bf hi = *(const v8bf*)(p + 8);
  return make16(lo, hi);
}

// B-fragment (32x16 bf16) from global matrix stored as M2[col][k] (row stride ldB).
// lane holds column = lane%16, 16 contiguous K starting at (lane<16 ? 0 : 16).
__device__ __forceinline__ v16bf frag_b_g(const bf16_t* __restrict__ B2, long ldB,
                                          int col, int kbase, int lane) {
  const bf16_t* p = B2 + (size_t)col * ldB + kbase + ((lane & 16) ? 16 : 0);
  v8bf lo = *(const v8bf*)p;
  v8bf hi = *(const v8bf*)(p + 8);
  return make16(lo, hi);
}

// ---------------------------------------------------------------------------
// Unified GEMM:  C[M x Ncols] (+=) A[M x Kdim] * B2[Ncols x Kdim]^T (+ bias)
// Block tile 128x128, 8 waves, each wave owns 32(M) x 64(N): 8 accumulators.
// bf16-A path: A tile staged by Tensor Data Mover, double-buffered.
// ---------------------------------------------------------------------------
template <bool A_F32, bool OUT_BF16, bool TRANS_OUT, bool HAS_BIAS, bool ATOMIC_OUT>
__global__ __launch_bounds__(256)
void gemm_bt_kernel(const void* __restrict__ Ap, const bf16_t* __restrict__ B2,
                    const float* __restrict__ bias, void* __restrict__ Cp,
                    int M, int Ncols, int Kdim, long ldA, long ldB, long ldC,
                    int kChunk)
{
  constexpr int NBUF = A_F32 ? 1 : 2;
  constexpr int BUFSZ = BT * LDS_STRIDE;
  __shared__ bf16_t ldsA[NBUF * BUFSZ];

  const int tid  = threadIdx.x;
  const int lane = tid & 31;
  const int w    = tid >> 5;
  const int wr   = w >> 1;   // 0..3 : 32-row strip
  const int wc   = w & 1;    // 0..1 : 64-col strip
  const int rowBlk = blockIdx.y * BT;
  const int colBlk = blockIdx.x * BT;
  int k0 = blockIdx.z * kChunk;
  int k1 = k0 + kChunk; if (k1 > Kdim) k1 = Kdim;

  v8f acc[2][4];
#pragma unroll
  for (int i = 0; i < 2; ++i)
#pragma unroll
    for (int j = 0; j < 4; ++j)
#pragma unroll
      for (int r = 0; r < 8; ++r) acc[i][j][r] = 0.f;

  // per-wave B columns (clamped; stores are guarded later)
  int cols[4];
#pragma unroll
  for (int et = 0; et < 4; ++et) {
    int c = colBlk + wc * 64 + et * 16 + (lane & 15);
    cols[et] = (c >= Ncols) ? (Ncols - 1) : c;
  }

  // one K-step of math: 2 A-frags from LDS, 4 prefetched B-frags, 8 WMMAs
  auto compute_step = [&](const bf16_t* abuf, int k) {
    v16bf a0 = frag_a_lds(abuf, wr * 32,      lane);
    v16bf a1 = frag_a_lds(abuf, wr * 32 + 16, lane);
    v16bf bfrag[4];
#pragma unroll
    for (int et = 0; et < 4; ++et)
      bfrag[et] = frag_b_g(B2, ldB, cols[et], k, lane);
#pragma unroll
    for (int et = 0; et < 4; ++et) {
      acc[0][et] = __builtin_amdgcn_wmma_f32_16x16x32_bf16(
          false, a0, false, bfrag[et], (short)0, acc[0][et], false, false);
      acc[1][et] = __builtin_amdgcn_wmma_f32_16x16x32_bf16(
          false, a1, false, bfrag[et], (short)0, acc[1][et], false, false);
    }
  };

  if (A_F32) {
    // ---- VALU staging with fused f32->bf16 convert ----
    const int sRow = tid >> 1;          // 0..127
    const int sCol = (tid & 1) * 16;    // 0 or 16
    for (int k = k0; k < k1; k += KSTEP) {
      {
        int gr = rowBlk + sRow;
        bool inR = (gr < M) && (k + sCol < Kdim);   // Kdim multiple of 16
        bf16_t* dst = &ldsA[sRow * LDS_STRIDE + sCol];
        v8bf lo, hi;
        if (inR) {
          const float4* s4 = (const float4*)((const float*)Ap + (size_t)gr * ldA + k + sCol);
          float4 f0 = s4[0], f1 = s4[1], f2 = s4[2], f3 = s4[3];
          lo[0]=(bf16_t)f0.x; lo[1]=(bf16_t)f0.y; lo[2]=(bf16_t)f0.z; lo[3]=(bf16_t)f0.w;
          lo[4]=(bf16_t)f1.x; lo[5]=(bf16_t)f1.y; lo[6]=(bf16_t)f1.z; lo[7]=(bf16_t)f1.w;
          hi[0]=(bf16_t)f2.x; hi[1]=(bf16_t)f2.y; hi[2]=(bf16_t)f2.z; hi[3]=(bf16_t)f2.w;
          hi[4]=(bf16_t)f3.x; hi[5]=(bf16_t)f3.y; hi[6]=(bf16_t)f3.z; hi[7]=(bf16_t)f3.w;
        } else {
#pragma unroll
          for (int i = 0; i < 8; ++i) { lo[i] = (bf16_t)0.f; hi[i] = (bf16_t)0.f; }
        }
        *(v8bf*)dst       = lo;
        *(v8bf*)(dst + 8) = hi;
      }
      __syncthreads();
      compute_step(ldsA, k);
      __syncthreads();
    }
  } else {
    // ---- Tensor Data Mover staging, double buffered (TENSORcnt) ----
    // D# (ISA 8.3/8.4): 2D tile 128 rows x 64B, row stride ldA elems,
    // data_size=2B; pad 4 DWORDs after every 16 DWORDs => LDS row stride 40 bf16.
    // tensor_dim0 = Kdim-k gives HW zero-fill on the K tail (OOB loads -> 0).
    auto issue_tdm = [&](int kk, int buf) {
      const bf16_t* gp = (const bf16_t*)Ap + (size_t)rowBlk * ldA + kk;
      unsigned long long ga = (unsigned long long)(size_t)gp;
      unsigned lds_off = (unsigned)(size_t)(bf16_t*)(ldsA + buf * BUFSZ); // low 32b = LDS offset
      unsigned rem = (unsigned)(Kdim - kk);
      u32x4 g0;
      g0[0] = 1u;                                  // count=1 (valid user D#)
      g0[1] = lds_off;                             // lds_addr
      g0[2] = (unsigned)ga;                        // global_addr[31:0]
      g0[3] = (unsigned)(ga >> 32) | (2u << 30);   // global_addr[56:32] | type=2
      i32x8 g1;
      g1[0] = (int)((1u << 16)       // data_size = 2B
                  | (1u << 20)       // pad_enable
                  | (3u << 22)       // pad_interval: every 16 DWORDs (64B)
                  | (3u << 25));     // pad_amount: 4 DWORDs (16B)
      g1[1] = (int)((rem & 0xffffu) << 16);              // tensor_dim0[15:0]
      g1[2] = (int)((rem >> 16) | ((unsigned)BT << 16)); // tensor_dim0[31:16] | tensor_dim1[15:0]
      g1[3] = (int)(32u << 16);                          // tensor_dim1 hi=0 | tile_dim0=32
      g1[4] = 128;                                       // tile_dim1=128, tile_dim2=0
      g1[5] = (int)(unsigned)ldA;                        // tensor_dim0_stride[31:0]
      g1[6] = (int)(((unsigned long long)ldA >> 32) & 0xffffu); // stride[47:32] | dim1_stride=0
      g1[7] = 0;
      i32x4 gz4; gz4[0] = 0; gz4[1] = 0; gz4[2] = 0; gz4[3] = 0;
      i32x8 gz8;
#pragma unroll
      for (int i = 0; i < 8; ++i) gz8[i] = 0;
      // clang-23 / therock-10.0 form: 6 args (g0, g1, g2, g3, g4, cpol)
      __builtin_amdgcn_tensor_load_to_lds(g0, g1, gz4, gz4, gz8, 0);
    };

    const int nsteps = (k1 - k0 + KSTEP - 1) / KSTEP;
    if (w == 0) issue_tdm(k0, 0);
    for (int s = 0; s < nsteps; ++s) {
      int k = k0 + s * KSTEP;
      if (w == 0) {
        if (s + 1 < nsteps) {
          issue_tdm(k + KSTEP, (s + 1) & 1);   // prefetch next tile
          __builtin_amdgcn_s_wait_tensorcnt(1); // in-order => tile s complete
        } else {
          __builtin_amdgcn_s_wait_tensorcnt(0);
        }
      }
      __syncthreads();
      compute_step(ldsA + (s & 1) * BUFSZ, k);
      __syncthreads();
    }
  }

  // ---- epilogue ----
#pragma unroll
  for (int mt = 0; mt < 2; ++mt) {
#pragma unroll
    for (int et = 0; et < 4; ++et) {
      int col  = colBlk + wc * 64 + et * 16 + (lane & 15);
      int rowb = rowBlk + wr * 32 + mt * 16 + ((lane & 16) ? 8 : 0);
      v8f a = acc[mt][et];
      if (HAS_BIAS) {
        float bv = (col < Ncols) ? bias[col] : 0.f;
#pragma unroll
        for (int r = 0; r < 8; ++r) a[r] += bv;
      }
      if (ATOMIC_OUT) {
        float* C = (float*)Cp;
        if (col < Ncols) {
#pragma unroll
          for (int r = 0; r < 8; ++r)
            if (rowb + r < M) atomicAdd(&C[(size_t)(rowb + r) * ldC + col], a[r]);
        }
      } else if (TRANS_OUT) {
        bf16_t* C = (bf16_t*)Cp;
        if (col < Ncols) {
          if (rowb + 8 <= M) {
            v8bf o;
#pragma unroll
            for (int r = 0; r < 8; ++r) o[r] = (bf16_t)a[r];
            *(v8bf*)(C + (size_t)col * ldC + rowb) = o;    // 16B store along M
          } else {
#pragma unroll
            for (int r = 0; r < 8; ++r)
              if (rowb + r < M) C[(size_t)col * ldC + rowb + r] = (bf16_t)a[r];
          }
        }
      } else if (OUT_BF16) {
        bf16_t* C = (bf16_t*)Cp;
        if (col < Ncols) {
#pragma unroll
          for (int r = 0; r < 8; ++r)
            if (rowb + r < M) C[(size_t)(rowb + r) * ldC + col] = (bf16_t)a[r];
        }
      } else {
        float* C = (float*)Cp;
        if (col < Ncols) {
#pragma unroll
          for (int r = 0; r < 8; ++r)
            if (rowb + r < M) C[(size_t)(rowb + r) * ldC + col] = a[r];
        }
      }
    }
  }
}

// ---------------------------------------------------------------------------
// helpers
// ---------------------------------------------------------------------------
__global__ void cvt_f32_bf16_kernel(const float* __restrict__ s, bf16_t* __restrict__ d, int n) {
  int i = blockIdx.x * 256 + threadIdx.x;
  if (i < n) d[i] = (bf16_t)s[i];
}

__global__ void zero_f32_kernel(float* p, int n) {
  int i = blockIdx.x * 256 + threadIdx.x;
  if (i < n) p[i] = 0.f;
}

// One block per row: max, then exp(s-m) -> bf16 (unnormalized), rowsum z.
__global__ void softmax_rows_kernel(const float* __restrict__ S, bf16_t* __restrict__ P,
                                    float* __restrict__ zrow, int N, long ld) {
  __shared__ float red[256];
  const int row = blockIdx.x;
  const float* s = S + (size_t)row * ld;
  float m = -3.4e38f;
  for (int i = threadIdx.x; i < N; i += 256) m = fmaxf(m, s[i]);
  red[threadIdx.x] = m; __syncthreads();
  for (int st = 128; st > 0; st >>= 1) {
    if (threadIdx.x < st) red[threadIdx.x] = fmaxf(red[threadIdx.x], red[threadIdx.x + st]);
    __syncthreads();
  }
  m = red[0]; __syncthreads();
  float z = 0.f;
  bf16_t* p = P + (size_t)row * ld;
  for (int i = threadIdx.x; i < N; i += 256) {
    float e = __expf(s[i] - m);
    z += e;
    p[i] = (bf16_t)e;
  }
  red[threadIdx.x] = z; __syncthreads();
  for (int st = 128; st > 0; st >>= 1) {
    if (threadIdx.x < st) red[threadIdx.x] += red[threadIdx.x + st];
    __syncthreads();
  }
  if (threadIdx.x == 0) zrow[row] = red[0];
}

__global__ void scale_out_kernel(const float* __restrict__ O, const float* __restrict__ z,
                                 float* __restrict__ out, int n) {
  int i = blockIdx.x * 256 + threadIdx.x;
  if (i < n) out[i] = O[i] / z[i >> 9];   // D = 512
}

// ---------------------------------------------------------------------------
extern "C" void kernel_launch(void* const* d_in, const int* in_sizes, int n_in,
                              void* d_out, int out_size, void* d_ws, size_t ws_size,
                              hipStream_t stream)
{
  constexpr int B = 512, D = 512, NE = 200000, NR = 10000;
  constexpr int KCH = 8192;   // split-K chunk for P@V

  const float* subj_q = (const float*)d_in[0];
  const float* rel_q  = (const float*)d_in[1];
  const float* obj_q  = (const float*)d_in[2];
  const float* ent    = (const float*)d_in[3];
  const float* relb   = (const float*)d_in[4];
  const float* wsrc[7] = { (const float*)d_in[5],  (const float*)d_in[7],
                           (const float*)d_in[9],  (const float*)d_in[11],
                           (const float*)d_in[13], (const float*)d_in[15],
                           (const float*)d_in[17] };
  const float* bq_s = (const float*)d_in[6];
  const float* bq_r = (const float*)d_in[8];
  const float* bq_o = (const float*)d_in[10];
  const float* bk_e = (const float*)d_in[12];
  const float* bv_e = (const float*)d_in[14];
  const float* bk_r = (const float*)d_in[16];
  const float* bv_r = (const float*)d_in[18];

  uint8_t* ws = (uint8_t*)d_ws;
  size_t off = 0;
  auto alloc = [&](size_t bytes) -> void* {
    void* p = ws + off;
    off += (bytes + 511) & ~(size_t)511;   // align + pad (covers 32B frag over-reads)
    return p;
  };

  bf16_t* Wb[7];
  for (int i = 0; i < 7; ++i) Wb[i] = (bf16_t*)alloc((size_t)D * D * 2);
  bf16_t* Qs    = (bf16_t*)alloc((size_t)B * D * 2);
  bf16_t* Qr    = (bf16_t*)alloc((size_t)B * D * 2);
  bf16_t* Qo    = (bf16_t*)alloc((size_t)B * D * 2);
  bf16_t* Kent  = (bf16_t*)alloc((size_t)NE * D * 2);
  bf16_t* Vtent = (bf16_t*)alloc((size_t)D * NE * 2);
  bf16_t* Krel  = (bf16_t*)alloc((size_t)NR * D * 2);
  bf16_t* Vtrel = (bf16_t*)alloc((size_t)D * NR * 2);
  float*  S     = (float*) alloc((size_t)B * NE * 4);
  bf16_t* P     = (bf16_t*)alloc((size_t)B * NE * 2);
  float*  zrow  = (float*) alloc((size_t)B * 4);
  float*  O     = (float*) alloc((size_t)B * D * 4);

  auto cdiv = [](int a, int b) { return (a + b - 1) / b; };

  // pass 0: weight conversion
  for (int i = 0; i < 7; ++i)
    cvt_f32_bf16_kernel<<<cdiv(D * D, 256), 256, 0, stream>>>(wsrc[i], Wb[i], D * D);

  // pass 1: projections (A f32, +bias, bf16 out; V banks stored transposed)
  auto projN = [&](const float* A, const bf16_t* W, const float* bias, bf16_t* C, int M) {
    gemm_bt_kernel<true, true, false, true, false>
        <<<dim3(cdiv(D, BT), cdiv(M, BT), 1), 256, 0, stream>>>(
            A, W, bias, C, M, D, D, D, D, D, D);
  };
  auto projT = [&](const float* A, const bf16_t* W, const float* bias, bf16_t* C, int M) {
    gemm_bt_kernel<true, true, true, true, false>
        <<<dim3(cdiv(D, BT), cdiv(M, BT), 1), 256, 0, stream>>>(
            A, W, bias, C, M, D, D, D, D, (long)M, D);
  };

  projN(subj_q, Wb[0], bq_s, Qs, B);
  projN(rel_q,  Wb[1], bq_r, Qr, B);
  projN(obj_q,  Wb[2], bq_o, Qo, B);
  projN(ent,  Wb[3], bk_e, Kent,  NE);
  projT(ent,  Wb[4], bv_e, Vtent, NE);
  projN(relb, Wb[5], bk_r, Krel,  NR);
  projT(relb, Wb[6], bv_r, Vtrel, NR);

  // pass 2: attention per set
  auto run_set = [&](const bf16_t* Q, const bf16_t* Kb, const bf16_t* Vt, int N,
                     float* outp) {
    // logits: S[B x N] = Q @ Kb^T (K = 512)
    gemm_bt_kernel<false, false, false, false, false>
        <<<dim3(cdiv(N, BT), cdiv(B, BT), 1), 256, 0, stream>>>(
            Q, Kb, nullptr, S, B, N, D, D, D, (long)N, D);
    // row softmax -> P (unnormalized exp, bf16), z[row]
    softmax_rows_kernel<<<B, 256, 0, stream>>>(S, P, zrow, N, (long)N);
    // O = P @ V  (split-K + f32 atomics)
    zero_f32_kernel<<<cdiv(B * D, 256), 256, 0, stream>>>(O, B * D);
    gemm_bt_kernel<false, false, false, false, true>
        <<<dim3(cdiv(D, BT), cdiv(B, BT), cdiv(N, KCH)), 256, 0, stream>>>(
            P, Vt, nullptr, O, B, D, N, (long)N, (long)N, D, KCH);
    // out = O / z
    scale_out_kernel<<<cdiv(B * D, 256), 256, 0, stream>>>(O, zrow, outp, B * D);
  };

  float* out = (float*)d_out;                      // [3, B, D]: subj, rel, obj
  run_set(Qs, Kent, Vtent, NE, out);
  run_set(Qr, Krel, Vtrel, NR, out + (size_t)B * D);
  run_set(Qo, Kent, Vtent, NE, out + 2 * (size_t)B * D);

  (void)in_sizes; (void)n_in; (void)out_size; (void)ws_size;
}
